// LSTM_16681652978185
// MI455X (gfx1250) — matrix operand order repaired and checked by reading, not compiled, not run on hardware
//
#include <hip/hip_runtime.h>

#define UNITS 4
#define BATCH 64
#define TLEN  256
#define FEAT  256
#define HID   256
#define KTOT  512   // HID + FEAT

typedef __attribute__((ext_vector_type(16))) __bf16 v16bf;
typedef __attribute__((ext_vector_type(8)))  float  v8f;
typedef __attribute__((ext_vector_type(4)))  unsigned int u32x4;
typedef __attribute__((ext_vector_type(8)))  int i32x8;
typedef __attribute__((ext_vector_type(4)))  int i32x4;

struct Frag32 { uint4 lo; uint4 hi; };   // 32 bytes = 16 bf16 = one WMMA A/B operand

#if __has_builtin(__builtin_amdgcn_tensor_load_to_lds) && __has_builtin(__builtin_amdgcn_s_wait_tensorcnt)
#define HAVE_TDM 1
#else
#define HAVE_TDM 0
#endif

__device__ __forceinline__ unsigned short f2bf(float x) {
    unsigned u = __builtin_bit_cast(unsigned, x);
    unsigned r = u + 0x7FFFu + ((u >> 16) & 1u);   // round-to-nearest-even
    return (unsigned short)(r >> 16);
}

__device__ __forceinline__ float fast_sigmoid(float x) {
    return 1.0f / (1.0f + __expf(-x));
}
__device__ __forceinline__ float fast_tanh(float x) {
    return 2.0f / (1.0f + __expf(-2.0f * x)) - 1.0f;
}

#if HAVE_TDM
typedef __attribute__((address_space(3))) unsigned short lds_us;
__device__ __forceinline__ unsigned lds_byte_off(const unsigned short* p) {
    return (unsigned)(unsigned long long)(lds_us*)p;
}
// TDM: DMA a 64x32 bf16 2-D tile (row stride = row_stride elements) into LDS,
// padding each 64B (16-DWORD) row with 4 DWORDs -> stride-40 bf16 rows in LDS.
// D# packing per CDNA5 ISA.
// 6-arg builtin form (clang-23 / therock-10.0): (g0, g1, g2, g3, g_extra, cpol).
__device__ __forceinline__ void tdm_load_tile_64x32(
    const unsigned short* g_tile, unsigned lds_addr, unsigned long long row_stride)
{
    unsigned long long ga = (unsigned long long)g_tile;
    u32x4 g0;
    g0[0] = 1u;                                                  // count=1 (valid user D#)
    g0[1] = lds_addr;                                            // lds_addr[31:0]
    g0[2] = (unsigned)(ga & 0xFFFFFFFFu);                        // global_addr[31:0]
    g0[3] = (unsigned)((ga >> 32) & 0x01FFFFFFu) | (2u << 30);   // global_addr[56:32] | type=2
    unsigned td0 = (unsigned)row_stride;                         // tensor_dim0 = full row length
    i32x8 g1;
    g1[0] = (1 << 16)                                            // data_size = 2 bytes
          | (1 << 20)                                            // pad_enable
          | (3 << 22)                                            // pad_interval = 16 DWORDs
          | (3 << 25);                                           // pad_amount   = 4 DWORDs
    g1[1] = (int)(td0 << 16);                                    // tensor_dim0[15:0]
    g1[2] = (int)((td0 >> 16) | ((unsigned)BATCH << 16));        // dim0[31:16] | tensor_dim1=64
    g1[3] = (int)(32u << 16);                                    // tile_dim0 = 32
    g1[4] = (int)BATCH;                                          // tile_dim1 = 64, tile_dim2 = 0
    g1[5] = (int)(row_stride & 0xFFFFFFFFull);                   // tensor_dim0_stride[31:0]
    g1[6] = (int)(row_stride >> 32);                             // stride[47:32]
    g1[7] = 0;
    i32x4 z4 = (i32x4)0;
    i32x8 z8 = (i32x8)0;
    __builtin_amdgcn_tensor_load_to_lds(g0, g1, z4, z4, z8, 0);
}
#endif

// ---------------------------------------------------------------------------
// One-time: gate weights (UNITS,512,256) f32 -> bf16 in WMMA B-fragment order:
// [u][g][kc(16)][nt(16)][lane(32)][8 dwords];  n = lane&15, kbase = (lane>>4)*16,
// dword j = (B[kbase+2j][n], B[kbase+2j+1][n]).
// ---------------------------------------------------------------------------
__global__ __launch_bounds__(256) void lstm_prep_w(
    const float* __restrict__ Wi, const float* __restrict__ Wf,
    const float* __restrict__ Wo, const float* __restrict__ Wc,
    unsigned int* __restrict__ Wfrag)
{
    unsigned q = blockIdx.x * 256u + threadIdx.x;   // 0 .. 1048575
    unsigned j    =  q        & 7u;
    unsigned lane = (q >> 3)  & 31u;
    unsigned nt   = (q >> 8)  & 15u;
    unsigned kc   = (q >> 12) & 15u;
    unsigned g    = (q >> 16) & 3u;
    unsigned u    = (q >> 18) & 3u;
    const float* W = (g == 0) ? Wi : (g == 1) ? Wf : (g == 2) ? Wo : Wc;
    unsigned k = kc * 32u + ((lane >> 4) << 4) + 2u * j;
    unsigned n = nt * 16u + (lane & 15u);
    float w0 = W[(u * KTOT + k) * HID + n];
    float w1 = W[(u * KTOT + k + 1) * HID + n];
    Wfrag[q] = (unsigned)f2bf(w0) | ((unsigned)f2bf(w1) << 16);
}

// One-time: X f32 -> bf16, same (B,T,FEAT) layout, 2 elems per thread.
__global__ __launch_bounds__(256) void lstm_prep_x(
    const float* __restrict__ X, unsigned int* __restrict__ Xbf)
{
    unsigned p = blockIdx.x * 256u + threadIdx.x;   // 0 .. 2097151
    float x0 = X[2u * p], x1 = X[2u * p + 1u];
    Xbf[p] = (unsigned)f2bf(x0) | ((unsigned)f2bf(x1) << 16);
}

// ---------------------------------------------------------------------------
// One LSTM timestep. Grid = 32 blocks: u = blk>>3, hid-tile(32) = blk&7.
// 256 threads = 8 waves; wave -> (gate = w>>1, 16-wide n subtile = w&1).
// A chunks double-buffered in LDS (TDM DMA when available), one barrier per
// K-chunk, B fragments pipelined in registers, A fragments gathered before
// the WMMA chain so ds_load latency overlaps the matrix ops.
// ---------------------------------------------------------------------------
__global__ __launch_bounds__(256) void lstm_step(
    const unsigned short* __restrict__ Xbf,     // [B][T][FEAT] bf16
    const unsigned short* __restrict__ Wfrag,   // fragment-ordered bf16
    const float* __restrict__ bI, const float* __restrict__ bF,
    const float* __restrict__ bO, const float* __restrict__ bC,
    const unsigned short* __restrict__ Hin,     // [U][B][HID] bf16 (prev step)
    unsigned short* __restrict__ Hout,          // [U][B][HID] bf16 (this step)
    float* __restrict__ Cst,                    // [U][B][HID] f32 cell state
    float* __restrict__ outSeq,                 // [B][T][U*HID]
    float* __restrict__ outH, float* __restrict__ outC,
    int t)
{
    const int u    = blockIdx.x >> 3;
    const int ht   = blockIdx.x & 7;            // 32-wide hid tile
    const int tid  = threadIdx.x;
    const int lane = tid & 31;
    const int wave = tid >> 5;
    const int g    = wave >> 1;                 // gate: 0=i 1=f 2=o 3=c
    const int nt2  = wave & 1;
    const int nt   = ht * 2 + nt2;

    __shared__ unsigned short As[2][64 * 40];   // double-buffered A chunk, 80B rows
    __shared__ float Gbuf[4][64][32];           // pre-activation gates

    v8f acc[4] = {};

    const int m_ld = tid >> 2;
    const int koff = (tid & 3) * 8;

    if (t == 0) {                               // H0 = 0: pre-zero both buffers once
        const uint4 z4 = make_uint4(0u, 0u, 0u, 0u);
        *reinterpret_cast<uint4*>(&As[0][m_ld * 40 + koff]) = z4;
        *reinterpret_cast<uint4*>(&As[1][m_ld * 40 + koff]) = z4;
    }

    auto loadB = [&](int kc) {
        const unsigned long long bidx =
            ((((unsigned long long)(u * 4 + g) * 16 + kc) * 16 + nt) * 32 + lane) * 16ull;
        Frag32 r;
        r.lo = *reinterpret_cast<const uint4*>(&Wfrag[bidx]);
        r.hi = *reinterpret_cast<const uint4*>(&Wfrag[bidx + 8]);
        return r;
    };

    auto stage = [&](int c) {                   // stage A chunk c into As[c&1]
        if (t == 0 && c < 8) return;            // zeros already staged
#if HAVE_TDM
        if (tid < 32) {                         // wave 0 issues one DMA per tile
            const unsigned short* src = (c < 8)
                ? &Hin[u * BATCH * HID + c * 32]
                : &Xbf[t * FEAT + (c * 32 - HID)];
            const unsigned long long stride = (c < 8) ? (unsigned long long)HID
                                                      : (unsigned long long)TLEN * FEAT;
            tdm_load_tile_64x32(src, lds_byte_off(&As[c & 1][0]), stride);
        }
#else
        const int kglob = c * 32 + koff;
        uint4 av;
        if (kglob < HID)
            av = *reinterpret_cast<const uint4*>(&Hin[(u * BATCH + m_ld) * HID + kglob]);
        else
            av = *reinterpret_cast<const uint4*>(&Xbf[(m_ld * TLEN + t) * FEAT + (kglob - HID)]);
        *reinterpret_cast<uint4*>(&As[c & 1][m_ld * 40 + koff]) = av;
#endif
    };

    auto finish = [&](int c) {                  // make chunk c visible to all waves
#if HAVE_TDM
        if (!(t == 0 && c < 8) && tid < 32)
            __builtin_amdgcn_s_wait_tensorcnt(0);
#endif
        __syncthreads();
    };

    Frag32 bnext = loadB(0);
    stage(0);
    finish(0);

    const int mrow  = lane & 15;
    const int khalf = (lane >> 4) * 16;

    for (int kc = 0; kc < 16; ++kc) {
        const Frag32 bcur = bnext;
        if (kc < 15) {
            bnext = loadB(kc + 1);              // pipeline next B fragment
            stage(kc + 1);                      // overlap next A DMA with compute
        }
        const v16bf bmat = __builtin_bit_cast(v16bf, bcur);
        const unsigned short* Ab = As[kc & 1];

        // Gather all 4 A fragments first (batched ds_load_b128)...
        Frag32 araw[4];
        #pragma unroll
        for (int mt = 0; mt < 4; ++mt) {
            const int m = mt * 16 + mrow;
            araw[mt].lo = *reinterpret_cast<const uint4*>(&Ab[m * 40 + khalf]);
            araw[mt].hi = *reinterpret_cast<const uint4*>(&Ab[m * 40 + 8 + khalf]);
        }
        // ...and pin them before the WMMA chain so the loads issue up front and
        // their s_wait_dscnt drains progressively behind earlier matrix ops.
        asm volatile("" ::: "memory");
        #pragma unroll
        for (int mt = 0; mt < 4; ++mt) {
            const v16bf amat = __builtin_bit_cast(v16bf, araw[mt]);
            acc[mt] = __builtin_amdgcn_wmma_f32_16x16x32_bf16(
                false, amat, false, bmat, (short)0, acc[mt], false, false);
        }
        if (kc < 15) finish(kc + 1);
    }

    // Spill pre-activation gate tiles to LDS (C/D layout: VGPR r -> M=r(+8), lane -> N).
    const int hi8  = (lane >> 4) * 8;
    const int ncol = nt2 * 16 + (lane & 15);
    #pragma unroll
    for (int mt = 0; mt < 4; ++mt) {
        #pragma unroll
        for (int r = 0; r < 8; ++r)
            Gbuf[g][mt * 16 + hi8 + r][ncol] = acc[mt][r];
    }
    __syncthreads();

    // Elementwise LSTM cell update for this (u, 64 batch, 32 hid) tile.
    const int m     = tid >> 2;
    const int nb    = (tid & 3) * 8;
    const int hbase = ht * 32;
    #pragma unroll
    for (int j = 0; j < 8; ++j) {
        const int n = nb + j;
        const int h = hbase + n;
        const float ipre = Gbuf[0][m][n] + bI[u * HID + h];
        const float fpre = Gbuf[1][m][n] + bF[u * HID + h];
        const float opre = Gbuf[2][m][n] + bO[u * HID + h];
        const float cpre = Gbuf[3][m][n] + bC[u * HID + h];
        const float I  = fast_sigmoid(ipre);
        const float F  = fast_sigmoid(fpre);
        const float O  = fast_sigmoid(opre);
        const float Cc = fast_tanh(cpre);
        const int cidx = (u * BATCH + m) * HID + h;
        const float cOld = (t == 0) ? 0.0f : Cst[cidx];
        const float cNew = F * cOld + I * Cc;
        const float hNew = O * fast_tanh(cNew);
        Cst[cidx]  = cNew;
        Hout[cidx] = f2bf(hNew);
        outSeq[((size_t)m * TLEN + t) * (UNITS * HID) + u * HID + h] = hNew;
        if (t == TLEN - 1) { outH[cidx] = hNew; outC[cidx] = cNew; }
    }
}

extern "C" void kernel_launch(void* const* d_in, const int* in_sizes, int n_in,
                              void* d_out, int out_size, void* d_ws, size_t ws_size,
                              hipStream_t stream) {
    const float* X  = (const float*)d_in[0];
    const float* Wi = (const float*)d_in[1];
    const float* bI = (const float*)d_in[2];
    const float* Wf = (const float*)d_in[3];
    const float* bF = (const float*)d_in[4];
    const float* Wo = (const float*)d_in[5];
    const float* bO = (const float*)d_in[6];
    const float* Wc = (const float*)d_in[7];
    const float* bC = (const float*)d_in[8];

    char* ws = (char*)d_ws;
    unsigned short* Wfrag = (unsigned short*)ws;                          // 4 MB
    unsigned short* Xbf   = (unsigned short*)(ws + (4ull << 20));         // 8 MB
    unsigned short* Hb0   = (unsigned short*)(ws + (12ull << 20));        // 128 KB
    unsigned short* Hb1   = (unsigned short*)(ws + (12ull << 20) + (128ull << 10));
    float*          Cst   = (float*)         (ws + (12ull << 20) + (256ull << 10)); // 256 KB

    float* outSeq = (float*)d_out;
    float* outH   = outSeq + (size_t)BATCH * TLEN * UNITS * HID;
    float* outC   = outH + (size_t)UNITS * BATCH * HID;

    lstm_prep_w<<<1048576 / 256, 256, 0, stream>>>(Wi, Wf, Wo, Wc, (unsigned int*)Wfrag);
    lstm_prep_x<<<2097152 / 256, 256, 0, stream>>>(X, (unsigned int*)Xbf);

    for (int t = 0; t < TLEN; ++t) {
        const unsigned short* Hin  = (t & 1) ? Hb1 : Hb0;
        unsigned short*       Hout = (t & 1) ? Hb0 : Hb1;
        lstm_step<<<32, 256, 0, stream>>>(Xbf, Wfrag, bI, bF, bO, bC,
                                          Hin, Hout, Cst, outSeq, outH, outC, t);
    }
}
